// TGN_63196148793567
// MI455X (gfx1250) — matrix operand order repaired and checked by reading, not compiled, not run on hardware
//
#include <hip/hip_runtime.h>
#include <hip/hip_bf16.h>

typedef __attribute__((ext_vector_type(16))) _Float16 v16h;
typedef __attribute__((ext_vector_type(8)))  _Float16 v8h;
typedef __attribute__((ext_vector_type(8)))  float    v8f;

#define N_NODE 200000
#define HID    100
#define EFEAT  172
#define TDIM   100
#define NEDGE  500000
#define MSG    472            // 2*HID + EFEAT + TDIM
#define MSG_P  480            // K padded: 15 tiles of 32
#define HID_P  128            // K padded: 4 tiles of 32
#define GATE_P 112            // per-gate output section, 7 tiles of 16
#define OUT_P  (3 * GATE_P)   // 336 padded outputs
#define NHB    7              // h-tile groups per gate
#define NODES_BLK 64          // 4 M-tiles per block
#define MSG_STRIDE 488        // LDS row stride (halves), conflict-free
#define GH_STRIDE  136

// ---------------- kernel 1: init workspace ----------------
__global__ void k_init(unsigned int* seg_bits, int* sel) {
    int i = blockIdx.x * blockDim.x + threadIdx.x;
    if (i < N_NODE) { seg_bits[i] = 0u; sel[i] = -1; }
}

// ---------------- kernel 2: segment-max timestamp per dst ----------------
// edge_ts in (1,3): positive floats -> uint bit compare == float compare
__global__ void k_segmax(const float* __restrict__ ets, const int* __restrict__ dst,
                         unsigned int* __restrict__ seg_bits) {
    int e = blockIdx.x * blockDim.x + threadIdx.x;
    if (e < NEDGE) atomicMax(&seg_bits[dst[e]], __float_as_uint(ets[e]));
}

// ---------------- kernel 3: pick max edge id among max-ts edges ----------------
__global__ void k_sel(const float* __restrict__ ets, const int* __restrict__ dst,
                      const unsigned int* __restrict__ seg_bits, int* __restrict__ sel) {
    int e = blockIdx.x * blockDim.x + threadIdx.x;
    if (e < NEDGE) {
        int d = dst[e];
        if (__float_as_uint(ets[e]) == seg_bits[d]) atomicMax(&sel[d], e);
    }
}

// ------- kernel 4: weights -> padded f16, gate-aligned rows (B friendly) -------
// row r in [0,336): gate g = r/112, within-gate output hh = r%112 (valid hh<100)
__global__ void k_wconv(const float* __restrict__ w_ih, const float* __restrict__ w_hh,
                        _Float16* __restrict__ wih2, _Float16* __restrict__ whh2) {
    int i = blockIdx.x * blockDim.x + threadIdx.x;
    if (i < OUT_P * MSG_P) {
        int r = i / MSG_P, c = i % MSG_P;
        int g = r / GATE_P, hh = r % GATE_P;
        float v = (hh < HID && c < MSG) ? w_ih[(g * HID + hh) * MSG + c] : 0.0f;
        wih2[i] = (_Float16)v;
    }
    if (i < OUT_P * HID_P) {
        int r = i / HID_P, c = i % HID_P;
        int g = r / GATE_P, hh = r % GATE_P;
        float v = (hh < HID && c < HID) ? w_hh[(g * HID + hh) * HID + c] : 0.0f;
        whh2[i] = (_Float16)v;
    }
}

// ---------------- kernel 5: message build + WMMA GRU (64 nodes/block) ----------------
__global__ void __launch_bounds__(128)
k_gru(const float* __restrict__ mem, const float* __restrict__ last_t,
      const float* __restrict__ efeat, const float* __restrict__ ets,
      const float* __restrict__ time_w, const float* __restrict__ time_b,
      const float* __restrict__ b_ih, const float* __restrict__ b_hh,
      const int* __restrict__ src,
      const unsigned int* __restrict__ seg_bits, const int* __restrict__ sel,
      const _Float16* __restrict__ wih2, const _Float16* __restrict__ whh2,
      float* __restrict__ out_mem, float* __restrict__ out_ts) {
    __shared__ _Float16 msgA[NODES_BLK * MSG_STRIDE];  // msg tiles, f16
    __shared__ _Float16 ghA [NODES_BLK * GH_STRIDE];   // memory tiles, f16
    __shared__ float deltaS[NODES_BLK], tsS[NODES_BLK];
    __shared__ int   selS[NODES_BLK], srcS[NODES_BLK], heS[NODES_BLK];

    const int tid   = threadIdx.x;
    const int lane  = tid & 31;
    const int wv    = tid >> 5;
    const int node0 = blockIdx.x * NODES_BLK;

    // ---- per-node scalar setup ----
    if (tid < NODES_BLK) {
        int node = node0 + tid;
        int s  = sel[node];
        int he = (s >= 0);
        int sc = he ? s : 0;
        int ss = src[sc];
        selS[tid] = sc;
        srcS[tid] = ss;
        heS[tid]  = he;
        tsS[tid]  = he ? __uint_as_float(seg_bits[node]) : 0.0f;
        deltaS[tid] = he ? (ets[sc] - last_t[ss]) : 0.0f;
    }
    __syncthreads();

    // ---- build msg A-tiles [64 x 480] f16 ----
    for (int idx = tid; idx < NODES_BLK * MSG_P; idx += 128) {
        int m = idx / MSG_P, c = idx % MSG_P;
        float v = 0.0f;
        if (heS[m]) {
            if (c < HID)                  v = mem[srcS[m] * HID + c];
            else if (c < 2 * HID)         v = mem[(node0 + m) * HID + (c - HID)];
            else if (c < 2 * HID + EFEAT) v = efeat[(long)selS[m] * EFEAT + (c - 2 * HID)];
            else if (c < MSG) {
                int j = c - (2 * HID + EFEAT);
                v = cosf(deltaS[m] * time_w[j] + time_b[j]);
            }
        }
        msgA[m * MSG_STRIDE + c] = (_Float16)v;
    }
    // ---- build memory A-tiles [64 x 128] f16 ----
    for (int idx = tid; idx < NODES_BLK * HID_P; idx += 128) {
        int m = idx / HID_P, c = idx % HID_P;
        float v = (c < HID) ? mem[(node0 + m) * HID + c] : 0.0f;
        ghA[m * GH_STRIDE + c] = (_Float16)v;
    }
    __syncthreads();

    const int ln15  = lane & 15;              // A:M (within tile), B:N, D:N
    const int kbA   = (lane < 16) ? 0 : 16;   // A-fragment K base
    const int kbB   = (lane < 16) ? 0 : 16;   // B-fragment K base
    const int mbase = (lane < 16) ? 0 : 8;    // D rows held by this lane

    // each wave owns h-tile groups hb = wv, wv+4 (7 total); wave-uniform loop
    for (int hb = wv; hb < NHB; hb += 4) {
        // acc[g][mt]: g=0 r (gi+gh), g=1 z (gi+gh), g=2 n (gi only)
        v8f acc[3][4];
        v8f accNH[4];                         // n-gate gh part, kept separate
#pragma unroll
        for (int g = 0; g < 3; ++g)
#pragma unroll
            for (int mt = 0; mt < 4; ++mt) { acc[g][mt] = (v8f){}; }
#pragma unroll
        for (int mt = 0; mt < 4; ++mt) accNH[mt] = (v8f){};

        // ---- gi: msg[64x480] x w_ih^T ----
        for (int kt = 0; kt < MSG_P / 32; ++kt) {
            int k0 = kt * 32;
            v16h a[4];
#pragma unroll
            for (int mt = 0; mt < 4; ++mt) {
                const _Float16* pa = &msgA[(mt * 16 + ln15) * MSG_STRIDE + k0 + kbA];
                v8h a0 = *(const v8h*)pa;
                v8h a1 = *(const v8h*)(pa + 8);
#pragma unroll
                for (int i = 0; i < 8; ++i) { a[mt][i] = a0[i]; a[mt][8 + i] = a1[i]; }
            }
#pragma unroll
            for (int g = 0; g < 3; ++g) {
                int row = (g * NHB + hb) * 16 + ln15;
                v16h b = *(const v16h*)&wih2[row * MSG_P + k0 + kbB];
#pragma unroll
                for (int mt = 0; mt < 4; ++mt)
                    acc[g][mt] = __builtin_amdgcn_wmma_f32_16x16x32_f16(
                        false, a[mt], false, b, (short)0, acc[g][mt], false, false);
            }
        }
        // ---- gh: memory[64x128] x w_hh^T (r,z join acc; n separate) ----
        for (int kt = 0; kt < HID_P / 32; ++kt) {
            int k0 = kt * 32;
            v16h a[4];
#pragma unroll
            for (int mt = 0; mt < 4; ++mt) {
                const _Float16* pa = &ghA[(mt * 16 + ln15) * GH_STRIDE + k0 + kbA];
                v8h a0 = *(const v8h*)pa;
                v8h a1 = *(const v8h*)(pa + 8);
#pragma unroll
                for (int i = 0; i < 8; ++i) { a[mt][i] = a0[i]; a[mt][8 + i] = a1[i]; }
            }
#pragma unroll
            for (int g = 0; g < 3; ++g) {
                int row = (g * NHB + hb) * 16 + ln15;
                v16h b = *(const v16h*)&whh2[row * HID_P + k0 + kbB];
                if (g < 2) {
#pragma unroll
                    for (int mt = 0; mt < 4; ++mt)
                        acc[g][mt] = __builtin_amdgcn_wmma_f32_16x16x32_f16(
                            false, a[mt], false, b, (short)0, acc[g][mt], false, false);
                } else {
#pragma unroll
                    for (int mt = 0; mt < 4; ++mt)
                        accNH[mt] = __builtin_amdgcn_wmma_f32_16x16x32_f16(
                            false, a[mt], false, b, (short)0, accNH[mt], false, false);
                }
            }
        }

        // ---- gate phase directly on D fragments (lane: column h, rows mbase+v) ----
        int h = hb * 16 + ln15;
        if (h < HID) {
            float brr = b_ih[h]           + b_hh[h];
            float bzz = b_ih[HID + h]     + b_hh[HID + h];
            float bin = b_ih[2 * HID + h];
            float bhn = b_hh[2 * HID + h];
#pragma unroll
            for (int mt = 0; mt < 4; ++mt) {
#pragma unroll
                for (int v = 0; v < 8; ++v) {
                    int node = node0 + mt * 16 + mbase + v;
                    float r = 1.0f / (1.0f + __expf(-(acc[0][mt][v] + brr)));
                    float z = 1.0f / (1.0f + __expf(-(acc[1][mt][v] + bzz)));
                    float n = tanhf(acc[2][mt][v] + bin + r * (accNH[mt][v] + bhn));
                    float mv = mem[node * HID + h];
                    out_mem[node * HID + h] = (1.0f - z) * n + z * mv;
                }
            }
        }
    }
    if (tid < NODES_BLK) out_ts[node0 + tid] = tsS[tid];
}

static inline size_t align256(size_t x) { return (x + 255) & ~(size_t)255; }

extern "C" void kernel_launch(void* const* d_in, const int* in_sizes, int n_in,
                              void* d_out, int out_size, void* d_ws, size_t ws_size,
                              hipStream_t stream) {
    const float* mem    = (const float*)d_in[0];
    const float* last_t = (const float*)d_in[1];
    const float* efeat  = (const float*)d_in[2];
    const float* ets    = (const float*)d_in[3];
    const float* time_w = (const float*)d_in[4];
    const float* time_b = (const float*)d_in[5];
    const float* w_ih   = (const float*)d_in[6];
    const float* w_hh   = (const float*)d_in[7];
    const float* b_ih   = (const float*)d_in[8];
    const float* b_hh   = (const float*)d_in[9];
    const int*   src    = (const int*)d_in[10];
    const int*   dst    = (const int*)d_in[11];

    char* ws = (char*)d_ws;
    unsigned int* seg_bits = (unsigned int*)ws; ws += align256((size_t)N_NODE * 4);
    int*          sel      = (int*)ws;          ws += align256((size_t)N_NODE * 4);
    _Float16*     wih2     = (_Float16*)ws;     ws += align256((size_t)OUT_P * MSG_P * 2);
    _Float16*     whh2     = (_Float16*)ws;     ws += align256((size_t)OUT_P * HID_P * 2);

    float* out_mem = (float*)d_out;
    float* out_ts  = out_mem + (size_t)N_NODE * HID;

    k_init  <<<(N_NODE + 255) / 256, 256, 0, stream>>>(seg_bits, sel);
    k_segmax<<<(NEDGE + 255) / 256, 256, 0, stream>>>(ets, dst, seg_bits);
    k_sel   <<<(NEDGE + 255) / 256, 256, 0, stream>>>(ets, dst, seg_bits, sel);
    k_wconv <<<(OUT_P * MSG_P + 255) / 256, 256, 0, stream>>>(w_ih, w_hh, wih2, whh2);
    k_gru   <<<N_NODE / NODES_BLK, 128, 0, stream>>>(mem, last_t, efeat, ets,
                                                     time_w, time_b, b_ih, b_hh,
                                                     src, seg_bits, sel,
                                                     wih2, whh2, out_mem, out_ts);
}
